// AttentionLayer_5995774345736
// MI455X (gfx1250) — matrix operand order repaired and checked by reading, not compiled, not run on hardware
//
#include <hip/hip_runtime.h>

typedef _Float16 half_t;
typedef __attribute__((ext_vector_type(16))) _Float16 v16h;
typedef __attribute__((ext_vector_type(8)))  float    v8f;
typedef unsigned int u32x4 __attribute__((ext_vector_type(4)));
typedef int          i32x8 __attribute__((ext_vector_type(8)));
typedef int          i32x4 __attribute__((ext_vector_type(4)));

#define D_MODEL 1024
#define N_BATCH 4
#define H_HEADS 16
#define E_DIM   64
#define L_SEQ   2048
#define M_ROWS  (N_BATCH * L_SEQ)   // 8192
#define NT_TILES (L_SEQ / 32)       // 64 S-tiles

#define BM 128
#define BN 64
#define BK 32

// LDS row strides (in half elements), padded for bank spread, 16B-aligned rows
#define XS_LD 40   // A tiles (k extent 32)
#define WT_LD 40   // transposed B tiles (k extent 32)
#define KT_LD 72   // K tile rows (e extent 64) -- matches TDM pad 4dw per 32dw
#define VT_LD 40   // transposed V tile rows (s extent 32)
#define PT_LD 40   // per-wave P tile rows (s extent 32)

// CDNA5 16-bit A-matrix 16x32 layout: lane holds row m=lane&15; half=lane>>4.
__device__ __forceinline__ int a_k_index(int h16, int i) {
    return h16 * 16 + 2 * i;
}

__device__ __forceinline__ v16h load_a_frag(const half_t* row, int h16) {
    v16h a;
    #pragma unroll
    for (int i = 0; i < 8; ++i) {
        int k = a_k_index(h16, i);
        a[2 * i]     = row[k];
        a[2 * i + 1] = row[k + 1];
    }
    return a;
}

// B layout: lane holds column n=lane&15, k values (lane>>4)*16 + j, j=0..15
__device__ __forceinline__ v16h load_b_frag(const half_t* p) {
    v16h b;
    #pragma unroll
    for (int j = 0; j < 16; ++j) b[j] = p[j];
    return b;
}

__device__ __forceinline__ v8f wmma_f16(v16h a, v16h b, v8f c) {
    return __builtin_amdgcn_wmma_f32_16x16x32_f16(false, a, false, b,
                                                  (short)0, c, false, false);
}

// ---------------------------------------------------------------------------
// QKV projection: out = (X @ W + bias) * scale, stored f16 head-major
// ---------------------------------------------------------------------------
__global__ __launch_bounds__(256)
void proj_qkv_kernel(const float* __restrict__ X, const float* __restrict__ W,
                     const float* __restrict__ bias, half_t* __restrict__ outH,
                     float out_scale)
{
    __shared__ half_t Xs[BM * XS_LD];
    __shared__ half_t Wt[BN * WT_LD];

    const int tid  = threadIdx.x;
    const int lane = tid & 31;
    const int wave = tid >> 5;
    const int ln   = lane & 15;
    const int h16  = lane >> 4;

    const int rowbase = blockIdx.x * BM;
    const int colbase = blockIdx.y * BN;

    v8f acc[4] = {};

    for (int kt = 0; kt < D_MODEL; kt += BK) {
        __syncthreads();
        { // X tile BMxBK fp32 -> f16 LDS (16 elems/thread)
            int r = tid >> 1;
            int c = (tid & 1) * 16;
            const float4* src =
                (const float4*)(X + (size_t)(rowbase + r) * D_MODEL + kt + c);
            #pragma unroll
            for (int q = 0; q < 4; ++q) {
                float4 v = src[q];
                half_t* dst = &Xs[r * XS_LD + c + q * 4];
                dst[0] = (half_t)v.x; dst[1] = (half_t)v.y;
                dst[2] = (half_t)v.z; dst[3] = (half_t)v.w;
            }
        }
        { // W tile BKxBN fp32 -> transposed f16 LDS Wt[n][k] (8 elems/thread)
            int r = tid >> 3;
            int c = (tid & 7) * 8;
            const float4* src =
                (const float4*)(W + (size_t)(kt + r) * D_MODEL + colbase + c);
            #pragma unroll
            for (int q = 0; q < 2; ++q) {
                float4 v = src[q];
                Wt[(c + q * 4 + 0) * WT_LD + r] = (half_t)v.x;
                Wt[(c + q * 4 + 1) * WT_LD + r] = (half_t)v.y;
                Wt[(c + q * 4 + 2) * WT_LD + r] = (half_t)v.z;
                Wt[(c + q * 4 + 3) * WT_LD + r] = (half_t)v.w;
            }
        }
        __syncthreads();

        v16h a = load_a_frag(&Xs[(wave * 16 + ln) * XS_LD], h16);
        #pragma unroll
        for (int t = 0; t < 4; ++t) {
            v16h b = load_b_frag(&Wt[(t * 16 + ln) * WT_LD + h16 * 16]);
            acc[t] = wmma_f16(a, b, acc[t]);
        }
    }

    #pragma unroll
    for (int t = 0; t < 4; ++t) {
        int col = colbase + t * 16 + ln;
        int h = col >> 6, e = col & 63;
        float bv = bias[col];
        #pragma unroll
        for (int r = 0; r < 8; ++r) {
            int row = rowbase + wave * 16 + r + 8 * h16;
            int n = row >> 11;
            int l = row & (L_SEQ - 1);
            float v = (acc[t][r] + bv) * out_scale;
            outH[(((size_t)(n * H_HEADS + h) * L_SEQ + l) << 6) + e] = (half_t)v;
        }
    }
}

// ---------------------------------------------------------------------------
// Flash attention per (n,h): grid (L_SEQ/128, N*H), 256 threads (8 waves)
// K tiles streamed by the Tensor Data Mover (double-buffered LDS, TDM pad
// reproduces the 72-half row stride); V tiles double-buffered via vector
// loads issued before the WMMA phase.
// ---------------------------------------------------------------------------
__global__ __launch_bounds__(256)
void attn_kernel(const half_t* __restrict__ Qh, const half_t* __restrict__ Kh,
                 const half_t* __restrict__ Vh, half_t* __restrict__ ctx)
{
    __shared__ half_t Kt[2][32 * KT_LD];
    __shared__ half_t Vt[2][64 * VT_LD];
    __shared__ half_t Pt[8][16 * PT_LD];

    const int tid = threadIdx.x, lane = tid & 31, wave = tid >> 5;
    const int ln = lane & 15, h16 = lane >> 4;

    const int nh    = blockIdx.y;
    const int qbase = blockIdx.x * 128;

    const half_t* Qp = Qh + (size_t)nh * L_SEQ * E_DIM;
    const half_t* Kp = Kh + (size_t)nh * L_SEQ * E_DIM;
    const half_t* Vp = Vh + (size_t)nh * L_SEQ * E_DIM;

    // Q fragments for the whole S loop (softmax scale folded into Q)
    const half_t* qrow = Qp + (size_t)(qbase + wave * 16 + ln) * E_DIM;
    v16h aq0, aq1;
    #pragma unroll
    for (int i = 0; i < 8; ++i) {
        int k = a_k_index(h16, i);
        aq0[2 * i] = qrow[k];          aq0[2 * i + 1] = qrow[k + 1];
        aq1[2 * i] = qrow[32 + k];     aq1[2 * i + 1] = qrow[32 + k + 1];
    }

    v8f o[4] = {};
    float mstat[8], lstat[8];
    #pragma unroll
    for (int r = 0; r < 8; ++r) { mstat[r] = -1e30f; lstat[r] = 0.0f; }

    const int ld_s = (tid * 8) >> 6;     // 0..31
    const int ld_e = (tid * 8) & 63;     // multiple of 8

    // TDM: 1-D contiguous 4KB K tile -> LDS, padded 4 dwords every 32 dwords
    auto tdm_load_k = [&](int tile, int b) {
        unsigned long long ga =
            (unsigned long long)(const void*)(Kp + (size_t)tile * (32 * E_DIM));
        unsigned int lds = (unsigned int)(size_t)(void*)&Kt[b][0];
        u32x4 g0;
        g0[0] = 1u;                                            // count=1
        g0[1] = lds;                                           // lds_addr
        g0[2] = (unsigned int)ga;                              // global_addr lo
        g0[3] = (unsigned int)((ga >> 32) & 0x1FFFFFFu) | (2u << 30); // hi|type=2
        i32x8 g1;
        g1[0] = (int)0x07110000u;  // data_size=2B, pad_en, interval=32dw, amt=4dw
        g1[1] = 0;                 // tensor_dim0 lo16 = 0 (dim0 = 0x100000)
        g1[2] = (int)((1u << 16) | 0x0010u);  // tensor_dim1=1 | tensor_dim0 hi16
        g1[3] = (int)(2048u << 16);           // tile_dim0 = 2048 elements
        g1[4] = 1;                            // tile_dim1 = 1
        g1[5] = 2048;                         // tensor_dim0_stride lo32
        g1[6] = 0;
        g1[7] = 0;
        i32x4 gz4 = {0, 0, 0, 0};
        i32x8 gz8 = {0, 0, 0, 0, 0, 0, 0, 0};
        __builtin_amdgcn_tensor_load_to_lds(g0, g1, gz4, gz4, gz8, 0);
    };
    auto load_v_regs = [&](int tile) -> uint4 {
        return *(const uint4*)(Vp + (size_t)tile * (32 * E_DIM) +
                               (size_t)ld_s * E_DIM + ld_e);
    };
    auto scatter_v = [&](uint4 v, int b) {
        union { uint4 u; half_t h[8]; } cv; cv.u = v;
        #pragma unroll
        for (int j = 0; j < 8; ++j) Vt[b][(ld_e + j) * VT_LD + ld_s] = cv.h[j];
    };

    // prologue: fill buffer 0
    if (wave == 0) tdm_load_k(0, 0);
    scatter_v(load_v_regs(0), 0);
    if (wave == 0) __builtin_amdgcn_s_wait_tensorcnt(0);
    __syncthreads();

    for (int i = 0; i < NT_TILES; ++i) {
        const int buf = i & 1, nbuf = buf ^ 1;
        const bool has_next = (i + 1 < NT_TILES);
        uint4 vreg = {};
        if (has_next) {
            if (wave == 0) tdm_load_k(i + 1, nbuf);   // TDM overlaps compute
            vreg = load_v_regs(i + 1);                // in flight during WMMAs
            if (i + 2 < NT_TILES)
                __builtin_prefetch(Vp + (size_t)(i + 2) * (32 * E_DIM) +
                                   (size_t)ld_s * E_DIM + ld_e, 0, 3);
        }

        // scores: two 16x16 C tiles, 4 WMMAs
        v8f c_lo = {}, c_hi = {};
        {
            const half_t* kb = Kt[buf];
            v16h b;
            b = load_b_frag(&kb[ln * KT_LD + h16 * 16]);
            c_lo = wmma_f16(aq0, b, c_lo);
            b = load_b_frag(&kb[ln * KT_LD + 32 + h16 * 16]);
            c_lo = wmma_f16(aq1, b, c_lo);
            b = load_b_frag(&kb[(16 + ln) * KT_LD + h16 * 16]);
            c_hi = wmma_f16(aq0, b, c_hi);
            b = load_b_frag(&kb[(16 + ln) * KT_LD + 32 + h16 * 16]);
            c_hi = wmma_f16(aq1, b, c_hi);
        }

        // online softmax (16-lane butterflies stay within half-wave rows)
        float alpha[8];
        #pragma unroll
        for (int r = 0; r < 8; ++r) {
            float t = fmaxf(c_lo[r], c_hi[r]);
            t = fmaxf(t, __shfl_xor(t, 1, 32));
            t = fmaxf(t, __shfl_xor(t, 2, 32));
            t = fmaxf(t, __shfl_xor(t, 4, 32));
            t = fmaxf(t, __shfl_xor(t, 8, 32));
            float mnew = fmaxf(mstat[r], t);
            alpha[r] = __expf(mstat[r] - mnew);
            mstat[r] = mnew;
            float plo = __expf(c_lo[r] - mnew);
            float phi = __expf(c_hi[r] - mnew);
            c_lo[r] = plo; c_hi[r] = phi;
            float rs = plo + phi;
            rs += __shfl_xor(rs, 1, 32);
            rs += __shfl_xor(rs, 2, 32);
            rs += __shfl_xor(rs, 4, 32);
            rs += __shfl_xor(rs, 8, 32);
            lstat[r] = lstat[r] * alpha[r] + rs;
        }
        #pragma unroll
        for (int t = 0; t < 4; ++t) {
            #pragma unroll
            for (int r = 0; r < 8; ++r) o[t][r] *= alpha[r];
        }

        // P (C layout) -> per-wave LDS -> A fragment (same-wave DS in-order)
        half_t* P = Pt[wave];
        #pragma unroll
        for (int r = 0; r < 8; ++r) {
            int m = r + 8 * h16;
            P[m * PT_LD + ln]      = (half_t)c_lo[r];
            P[m * PT_LD + 16 + ln] = (half_t)c_hi[r];
        }
        v16h ap = load_a_frag(&Pt[wave][ln * PT_LD], h16);

        // O += P @ V : 4 WMMAs
        #pragma unroll
        for (int t = 0; t < 4; ++t) {
            v16h bv = load_b_frag(&Vt[buf][(t * 16 + ln) * VT_LD + h16 * 16]);
            o[t] = wmma_f16(ap, bv, o[t]);
        }

        if (has_next) {
            scatter_v(vreg, nbuf);
            if (wave == 0) __builtin_amdgcn_s_wait_tensorcnt(0);
            __syncthreads();   // next tile visible; old buffer safe to overwrite
        }
    }

    // epilogue: O / l -> ctx [n][l][h*64+e] f16
    const int n = nh >> 4, hh = nh & 15;
    #pragma unroll
    for (int r = 0; r < 8; ++r) {
        float inv = 1.0f / lstat[r];
        size_t row = (size_t)n * L_SEQ + qbase + wave * 16 + r + 8 * h16;
        #pragma unroll
        for (int t = 0; t < 4; ++t) {
            int col = hh * 64 + t * 16 + ln;
            ctx[row * D_MODEL + col] = (half_t)(o[t][r] * inv);
        }
    }
}

// ---------------------------------------------------------------------------
// Output projection: out = ctx(f16) @ Wo + bo  -> fp32
// ---------------------------------------------------------------------------
__global__ __launch_bounds__(256)
void out_proj_kernel(const half_t* __restrict__ Xh, const float* __restrict__ W,
                     const float* __restrict__ bias, float* __restrict__ out)
{
    __shared__ half_t Xs[BM * XS_LD];
    __shared__ half_t Wt[BN * WT_LD];

    const int tid  = threadIdx.x;
    const int lane = tid & 31;
    const int wave = tid >> 5;
    const int ln   = lane & 15;
    const int h16  = lane >> 4;

    const int rowbase = blockIdx.x * BM;
    const int colbase = blockIdx.y * BN;

    v8f acc[4] = {};

    for (int kt = 0; kt < D_MODEL; kt += BK) {
        __syncthreads();
        {
            int r = tid >> 1;
            int c = (tid & 1) * 16;
            const uint4* src =
                (const uint4*)(Xh + (size_t)(rowbase + r) * D_MODEL + kt + c);
            uint4 v0 = src[0], v1 = src[1];
            *(uint4*)&Xs[r * XS_LD + c]     = v0;
            *(uint4*)&Xs[r * XS_LD + c + 8] = v1;
        }
        {
            int r = tid >> 3;
            int c = (tid & 7) * 8;
            const float4* src =
                (const float4*)(W + (size_t)(kt + r) * D_MODEL + colbase + c);
            #pragma unroll
            for (int q = 0; q < 2; ++q) {
                float4 v = src[q];
                Wt[(c + q * 4 + 0) * WT_LD + r] = (half_t)v.x;
                Wt[(c + q * 4 + 1) * WT_LD + r] = (half_t)v.y;
                Wt[(c + q * 4 + 2) * WT_LD + r] = (half_t)v.z;
                Wt[(c + q * 4 + 3) * WT_LD + r] = (half_t)v.w;
            }
        }
        __syncthreads();

        v16h a = load_a_frag(&Xs[(wave * 16 + ln) * XS_LD], h16);
        #pragma unroll
        for (int t = 0; t < 4; ++t) {
            v16h b = load_b_frag(&Wt[(t * 16 + ln) * WT_LD + h16 * 16]);
            acc[t] = wmma_f16(a, b, acc[t]);
        }
    }

    #pragma unroll
    for (int t = 0; t < 4; ++t) {
        int col = colbase + t * 16 + ln;
        float bv = bias[col];
        #pragma unroll
        for (int r = 0; r < 8; ++r) {
            int row = rowbase + wave * 16 + r + 8 * h16;
            out[(size_t)row * D_MODEL + col] = acc[t][r] + bv;
        }
    }
}

extern "C" void kernel_launch(void* const* d_in, const int* in_sizes, int n_in,
                              void* d_out, int out_size, void* d_ws, size_t ws_size,
                              hipStream_t stream) {
    (void)in_sizes; (void)n_in; (void)out_size; (void)ws_size;

    const float* queries = (const float*)d_in[0];
    const float* keys    = (const float*)d_in[1];
    const float* values  = (const float*)d_in[2];
    const float* Wq = (const float*)d_in[3];
    const float* bq = (const float*)d_in[4];
    const float* Wk = (const float*)d_in[5];
    const float* bk = (const float*)d_in[6];
    const float* Wv = (const float*)d_in[7];
    const float* bv = (const float*)d_in[8];
    const float* Wo = (const float*)d_in[9];
    const float* bo = (const float*)d_in[10];
    float* out = (float*)d_out;

    const size_t elems = (size_t)M_ROWS * D_MODEL;
    half_t* Qh  = (half_t*)d_ws;
    half_t* Kh  = Qh + elems;
    half_t* Vh  = Kh + elems;
    half_t* ctx = Vh + elems;       // 4 x 16 MB = 64 MB workspace

    dim3 gproj(M_ROWS / BM, D_MODEL / BN);
    proj_qkv_kernel<<<gproj, 256, 0, stream>>>(queries, Wq, bq, Qh, 0.125f);
    proj_qkv_kernel<<<gproj, 256, 0, stream>>>(keys,    Wk, bk, Kh, 1.0f);
    proj_qkv_kernel<<<gproj, 256, 0, stream>>>(values,  Wv, bv, Vh, 1.0f);

    dim3 gattn(L_SEQ / 128, N_BATCH * H_HEADS);
    attn_kernel<<<gattn, 256, 0, stream>>>(Qh, Kh, Vh, ctx);

    out_proj_kernel<<<gproj, 256, 0, stream>>>(ctx, Wo, bo, out);
}
